// HierarchicalGNNBlock_3848290697352
// MI455X (gfx1250) — matrix operand order repaired and checked by reading, not compiled, not run on hardware
//
#include <hip/hip_runtime.h>
#include <hip/hip_bf16.h>
#include <hip/hip_fp16.h>

// ---------------------------------------------------------------------------
// HierarchicalGNNBlock forward for MI455X (gfx1250, wave32, WMMA)
// Heavy MLPs -> v_wmma_f32_16x16x32_f16 tiled GEMM + fused LayerNorm/ReLU.
// Graph segment-sums -> global f32 atomics. All scratch carved from d_ws.
// ---------------------------------------------------------------------------

#define NNODE  50000
#define NEDGE  250000
#define NBIP   250000
#define NSE    10000
#define NCLU   1000
#define LATENT 128
#define HIDDEN 256
#define EMB    16

typedef __attribute__((ext_vector_type(16))) _Float16 v16h;
typedef __attribute__((ext_vector_type(8)))  float    v8f;
typedef _Float16 half_t;

static inline unsigned gb(size_t n, unsigned t = 256) {
    size_t b = (n + t - 1) / t;
    return (unsigned)(b ? b : 1);
}
static inline int cdiv(int a, int b) { return (a + b - 1) / b; }

// ---------------------------------------------------------------------------
// WMMA GEMM:  Y[M x NOUT] = act(LN(X[M x K] @ W[K x NOUT] + bias))  (+residual)
// X is f16 row-major. Wt is f16 *transposed* [NOUT x K]. Accum f32 via WMMA.
// Block = 128 threads = 4 waves; each wave owns a 16-row stripe of a 64-row
// block tile. LayerNorm is reduced per-row with wave32 shuffles.
// ---------------------------------------------------------------------------
template <int K, int NOUT>
__global__ __launch_bounds__(128) void wmma_mlp_layer(
    const half_t* __restrict__ X, const half_t* __restrict__ Wt,
    const float* __restrict__ bias, const float* __restrict__ gamma,
    const float* __restrict__ beta, const float* __restrict__ residual,
    half_t* __restrict__ Yh, float* __restrict__ Yf, int M, int doLN) {
    extern __shared__ float sh[];  // 64 * NOUT floats

    const int lane = threadIdx.x & 31;
    const int wave = threadIdx.x >> 5;
    const int lrow = wave * 16;              // wave's base row inside block tile
    const int blockRow = blockIdx.x * 64;    // block tile's first global row
    const int l15 = lane & 15;
    const bool hi = lane >= 16;

    // A-matrix source row for this lane (clamped: padding rows never stored).
    int arow = blockRow + lrow + l15;
    if (arow >= M) arow = M - 1;
    const half_t* xrow = X + (size_t)arow * K;
    __builtin_prefetch(xrow, 0, 3);  // global_prefetch_b8

    for (int nb = 0; nb < NOUT; nb += 16) {
        // B column n = nb + l15; lanes >=16 supply K {8..15, 24..31} of each chunk
        const half_t* wrow = Wt + (size_t)(nb + l15) * K + (hi ? 8 : 0);
        v8f acc = {};
#pragma unroll
        for (int k0 = 0; k0 < K; k0 += 32) {
            union { v16h v; uint4 u[2]; } a, b;
            // A 16x32 f16 layout: lanes 0-15 -> K {k0..k0+7, k0+16..k0+23},
            //                     lanes 16-31 -> K {k0+8..k0+15, k0+24..k0+31}
            const int ab = k0 + (hi ? 8 : 0);
            a.u[0] = *reinterpret_cast<const uint4*>(xrow + ab);
            a.u[1] = *reinterpret_cast<const uint4*>(xrow + ab + 16);
            // B 32x16 f16 layout: lane = column, K order same as A
            b.u[0] = *reinterpret_cast<const uint4*>(wrow + k0);
            b.u[1] = *reinterpret_cast<const uint4*>(wrow + k0 + 16);
            acc = __builtin_amdgcn_wmma_f32_16x16x32_f16(
                false, a.v, false, b.v, (short)0, acc, false, false);
        }
        // C/D layout: acc[r] -> row (r + hi*8), col l15
        const float bc = bias[nb + l15];
        const int rbase = lrow + (hi ? 8 : 0);
#pragma unroll
        for (int r = 0; r < 8; ++r)
            sh[(rbase + r) * NOUT + nb + l15] = acc[r] + bc;
    }
    __syncthreads();

    // Fused LayerNorm + ReLU (+residual) per row, wave32 shuffle reduction.
    for (int r = 0; r < 16; ++r) {
        const int row = lrow + r;
        float s1 = 0.f, s2 = 0.f;
        for (int n = lane; n < NOUT; n += 32) {
            float x = sh[row * NOUT + n];
            s1 += x;
            s2 += x * x;
        }
#pragma unroll
        for (int off = 16; off > 0; off >>= 1) {
            s1 += __shfl_xor(s1, off, 32);
            s2 += __shfl_xor(s2, off, 32);
        }
        const float mu = s1 * (1.f / NOUT);
        const float inv =
            rsqrtf(fmaxf(s2 * (1.f / NOUT) - mu * mu, 0.f) + 1e-5f);
        const int grow = blockRow + row;
        if (grow < M) {
            for (int n = lane; n < NOUT; n += 32) {
                float y = sh[row * NOUT + n];
                if (doLN) {
                    y = (y - mu) * inv * gamma[n] + beta[n];
                    y = fmaxf(y, 0.f);
                }
                if (residual) y += residual[(size_t)grow * NOUT + n];
                if (Yf) Yf[(size_t)grow * NOUT + n] = y;
                if (Yh) Yh[(size_t)grow * NOUT + n] = (half_t)y;
            }
        }
    }
}

template <int K, int NOUT>
static void launch_layer(const half_t* X, const half_t* Wt, const float* b,
                         const float* g, const float* be, const float* res,
                         half_t* Yh, float* Yf, int M, int doLN,
                         hipStream_t s) {
    wmma_mlp_layer<K, NOUT><<<cdiv(M, 64), 128, (size_t)64 * NOUT * sizeof(float), s>>>(
        X, Wt, b, g, be, res, Yh, Yf, M, doLN);
}

// ---------------------------------------------------------------------------
// Elementwise / scatter helpers
// ---------------------------------------------------------------------------
__global__ void k_zero(float* __restrict__ p, size_t n) {
    for (size_t i = blockIdx.x * (size_t)blockDim.x + threadIdx.x; i < n;
         i += (size_t)gridDim.x * blockDim.x)
        p[i] = 0.f;
}

__global__ void k_f2h(const float* __restrict__ a, half_t* __restrict__ o,
                      size_t n) {
    for (size_t i = blockIdx.x * (size_t)blockDim.x + threadIdx.x; i < n;
         i += (size_t)gridDim.x * blockDim.x)
        o[i] = (half_t)a[i];
}

// W [K x NOUT] f32 -> Wt [NOUT x K] f16 (transposed for contiguous B loads)
__global__ void k_convert_wt(const float* __restrict__ W,
                             half_t* __restrict__ Wt, int K, int NOUT) {
    size_t n = (size_t)K * NOUT;
    for (size_t i = blockIdx.x * (size_t)blockDim.x + threadIdx.x; i < n;
         i += (size_t)gridDim.x * blockDim.x) {
        int k = (int)(i / NOUT), c = (int)(i % NOUT);
        Wt[(size_t)c * K + k] = (half_t)W[i];
    }
}

__global__ void k_scatter_emb(const float* __restrict__ emb,
                              const int* __restrict__ cl,
                              float* __restrict__ sums,
                              float* __restrict__ cnt, size_t n) {
    for (size_t i = blockIdx.x * (size_t)blockDim.x + threadIdx.x; i < n;
         i += (size_t)gridDim.x * blockDim.x) {
        int node = (int)(i / EMB), d = (int)(i % EMB);
        int c = cl[node];
        atomicAdd(&sums[(size_t)c * EMB + d], emb[i]);
        if (d == 0) atomicAdd(&cnt[c], 1.f);
    }
}

__global__ void k_finalize_means(float* __restrict__ m,
                                 const float* __restrict__ cnt, int C) {
    int c = blockIdx.x * blockDim.x + threadIdx.x;
    if (c >= C) return;
    float inv = 1.f / fmaxf(cnt[c], 1.f);
    float v[EMB], s = 0.f;
#pragma unroll
    for (int d = 0; d < EMB; ++d) {
        v[d] = m[(size_t)c * EMB + d] * inv;
        s += v[d] * v[d];
    }
    float r = 1.f / fmaxf(sqrtf(s), 1e-12f);
#pragma unroll
    for (int d = 0; d < EMB; ++d) m[(size_t)c * EMB + d] = v[d] * r;
}

__global__ void k_edge_dot(const float* __restrict__ A,
                           const float* __restrict__ B,
                           const int* __restrict__ s,
                           const int* __restrict__ d, float* __restrict__ l,
                           int E) {
    for (int e = blockIdx.x * blockDim.x + threadIdx.x; e < E;
         e += gridDim.x * blockDim.x) {
        const float* a = A + (size_t)s[e] * EMB;
        const float* b = B + (size_t)d[e] * EMB;
        float acc = 0.f;
#pragma unroll
        for (int k = 0; k < EMB; ++k) acc = fmaf(a[k], b[k], acc);
        l[e] = acc;
    }
}

__global__ void k_stats(const float* __restrict__ l, int n,
                        float* __restrict__ st) {
    float s1 = 0.f, s2 = 0.f;
    for (int i = blockIdx.x * blockDim.x + threadIdx.x; i < n;
         i += gridDim.x * blockDim.x) {
        float x = l[i];
        s1 += x;
        s2 += x * x;
    }
#pragma unroll
    for (int off = 16; off > 0; off >>= 1) {
        s1 += __shfl_xor(s1, off, 32);
        s2 += __shfl_xor(s2, off, 32);
    }
    if ((threadIdx.x & 31) == 0) {
        atomicAdd(&st[0], s1);
        atomicAdd(&st[1], s2);
    }
}

// BatchNorm1d(1) -> sigmoid/exp -> atomic dst-sum
__global__ void k_edge_w(const float* __restrict__ l,
                         const float* __restrict__ st,
                         const float* __restrict__ g,
                         const float* __restrict__ b, int expMode,
                         float* __restrict__ w, float* __restrict__ ssum,
                         const int* __restrict__ dst, int n) {
    const float invn = 1.f / (float)n;
    const float mu = st[0] * invn;
    const float var = st[1] * invn - mu * mu;
    const float isd = rsqrtf(var + 1e-5f);
    const float gg = g[0], bb = b[0];
    for (int e = blockIdx.x * blockDim.x + threadIdx.x; e < n;
         e += gridDim.x * blockDim.x) {
        float x = (l[e] - mu) * isd * gg + bb;
        float ww = expMode ? expf(x) : 1.f / (1.f + expf(-x));
        w[e] = ww;
        atomicAdd(&ssum[dst[e]], ww);
    }
}

__global__ void k_edge_norm(float* __restrict__ w,
                            const float* __restrict__ ssum,
                            const int* __restrict__ dst, int n) {
    for (int e = blockIdx.x * blockDim.x + threadIdx.x; e < n;
         e += gridDim.x * blockDim.x)
        w[e] = w[e] / (ssum[dst[e]] + 1e-12f);
}

// out[sidx[e]*D+d] += vals[(gidx?gidx[e]:e)*D+d] * (w?w[e]:1)
__global__ void k_sgw_add(const float* __restrict__ vals,
                          const int* __restrict__ gidx,
                          const int* __restrict__ sidx,
                          const float* __restrict__ w, float* __restrict__ out,
                          int E, int D) {
    size_t n = (size_t)E * D;
    for (size_t i = blockIdx.x * (size_t)blockDim.x + threadIdx.x; i < n;
         i += (size_t)gridDim.x * blockDim.x) {
        int e = (int)(i / D), d = (int)(i % D);
        int src = gidx ? gidx[e] : e;
        float v = vals[(size_t)src * D + d];
        if (w) v *= w[e];
        atomicAdd(&out[(size_t)sidx[e] * D + d], v);
    }
}

// X[r] = f16([a[r] | b[r] | c[r]]), each LATENT wide
__global__ void k_concat3h(const float* __restrict__ a,
                           const float* __restrict__ b,
                           const float* __restrict__ c,
                           half_t* __restrict__ X, int M) {
    size_t n = (size_t)M * (3 * LATENT);
    for (size_t i = blockIdx.x * (size_t)blockDim.x + threadIdx.x; i < n;
         i += (size_t)gridDim.x * blockDim.x) {
        int r = (int)(i / (3 * LATENT)), col = (int)(i % (3 * LATENT));
        float v;
        if (col < LATENT) v = a[(size_t)r * LATENT + col];
        else if (col < 2 * LATENT) v = b[(size_t)r * LATENT + col - LATENT];
        else v = c[(size_t)r * LATENT + col - 2 * LATENT];
        X[i] = (half_t)v;
    }
}

// X[r] = f16([base[i0[r]] | base[i1[r]] | (extra? extra[r])]), widths D; W total
__global__ void k_gather2cat1h(const float* __restrict__ base,
                               const int* __restrict__ i0,
                               const int* __restrict__ i1,
                               const float* __restrict__ extra,
                               half_t* __restrict__ X, int M, int D, int W) {
    size_t n = (size_t)M * W;
    for (size_t i = blockIdx.x * (size_t)blockDim.x + threadIdx.x; i < n;
         i += (size_t)gridDim.x * blockDim.x) {
        int r = (int)(i / W), col = (int)(i % W);
        float v;
        if (col < D) v = base[(size_t)i0[r] * D + col];
        else if (col < 2 * D) v = base[(size_t)i1[r] * D + col - D];
        else v = extra[(size_t)r * D + col - 2 * D];
        X[i] = (half_t)v;
    }
}

// supernodes = [means(16) | enc(112)] -> 128 f32
__global__ void k_concat2f(const float* __restrict__ a,
                           const float* __restrict__ b, float* __restrict__ o,
                           int C) {
    size_t n = (size_t)C * LATENT;
    for (size_t i = blockIdx.x * (size_t)blockDim.x + threadIdx.x; i < n;
         i += (size_t)gridDim.x * blockDim.x) {
        int r = (int)(i / LATENT), col = (int)(i % LATENT);
        o[i] = (col < EMB) ? a[(size_t)r * EMB + col]
                           : b[(size_t)r * (LATENT - EMB) + col - EMB];
    }
}

__global__ void k_l2norm_rows(float* __restrict__ Y, int M, int D) {
    int r = blockIdx.x * blockDim.x + threadIdx.x;
    if (r >= M) return;
    float s = 0.f;
    for (int d = 0; d < D; ++d) {
        float v = Y[(size_t)r * D + d];
        s += v * v;
    }
    float inv = 1.f / fmaxf(sqrtf(s), 1e-12f);
    for (int d = 0; d < D; ++d) Y[(size_t)r * D + d] *= inv;
}

__global__ void k_copy_i32(const int* __restrict__ a, int* __restrict__ o,
                           int n) {
    for (int i = blockIdx.x * blockDim.x + threadIdx.x; i < n;
         i += gridDim.x * blockDim.x)
        o[i] = a[i];
}

// ---------------------------------------------------------------------------
// Host orchestration
// ---------------------------------------------------------------------------
extern "C" void kernel_launch(void* const* d_in, const int* in_sizes, int n_in,
                              void* d_out, int out_size, void* d_ws,
                              size_t ws_size, hipStream_t stream) {
    (void)in_sizes; (void)n_in; (void)out_size; (void)ws_size;

    // --- input map (JAX pytree: dicts flattened in sorted-key order) -------
    const float* embeddings = (const float*)d_in[0];
    const float* nodes_in   = (const float*)d_in[1];
    const float* edges_in   = (const float*)d_in[2];
    const float* bip_b = (const float*)d_in[3];   // bip_bn.b
    const float* bip_g = (const float*)d_in[4];   // bip_bn.g

    struct MlpP { const float *W0, *b0, *be0, *g0, *W1, *b1, *be1, *g1; };
    auto mlpAt = [&](int i) {
        MlpP m{(const float*)d_in[i + 0], (const float*)d_in[i + 1],
               (const float*)d_in[i + 2], (const float*)d_in[i + 3],
               (const float*)d_in[i + 4], (const float*)d_in[i + 5],
               (const float*)d_in[i + 6], (const float*)d_in[i + 7]};
        return m;
    };
    // cells[ci]: keys sorted -> edge, node, superedge, supernode (8 leaves each)
    MlpP c_edge[2], c_node[2], c_sedge[2], c_snode[2];
    for (int ci = 0; ci < 2; ++ci) {
        int b0 = 5 + ci * 32;
        c_edge[ci]  = mlpAt(b0 + 0);
        c_node[ci]  = mlpAt(b0 + 8);
        c_sedge[ci] = mlpAt(b0 + 16);
        c_snode[ci] = mlpAt(b0 + 24);
    }
    MlpP outp = mlpAt(69);
    const float* sup_b = (const float*)d_in[77];  // super_bn.b
    const float* sup_g = (const float*)d_in[78];  // super_bn.g
    MlpP seenc = mlpAt(79);                       // superedge_encoder
    MlpP snenc = mlpAt(87);                       // supernode_encoder
    const int* graph0 = (const int*)d_in[95];
    const int* graph1 = graph0 + NEDGE;
    const int* bsrc = (const int*)d_in[96];
    const int* bdst = (const int*)d_in[97];
    const int* sg0 = (const int*)d_in[98];
    const int* sg1 = sg0 + NSE;
    const int* clusters = (const int*)d_in[99];

    // --- workspace bump allocator ------------------------------------------
    char* wsp = (char*)d_ws;
    size_t off = 0;
    auto alloc = [&](size_t bytes) -> void* {
        void* p = wsp + off;
        off += (bytes + 255) & ~(size_t)255;
        return p;
    };
    struct WtH { half_t *W0, *W1; };
    auto allocW = [&](int K, int H, int O) {
        WtH w;
        w.W0 = (half_t*)alloc((size_t)K * H * sizeof(half_t));
        w.W1 = (half_t*)alloc((size_t)H * O * sizeof(half_t));
        return w;
    };
    WtH w_snenc = allocW(LATENT, HIDDEN, LATENT - EMB);
    WtH w_seenc = allocW(2 * LATENT, HIDDEN, LATENT);
    WtH w_out   = allocW(LATENT, HIDDEN, EMB);
    WtH w_cedge[2], w_cnode[2], w_csedge[2], w_csnode[2];
    for (int ci = 0; ci < 2; ++ci) {
        w_cedge[ci]  = allocW(3 * LATENT, HIDDEN, LATENT);
        w_cnode[ci]  = allocW(3 * LATENT, HIDDEN, LATENT);
        w_csedge[ci] = allocW(3 * LATENT, HIDDEN, LATENT);
        w_csnode[ci] = allocW(3 * LATENT, HIDDEN, LATENT);
    }

    float* means  = (float*)alloc((size_t)NCLU * EMB * 4);
    float* cnt    = (float*)alloc((size_t)NCLU * 4);
    float* stats  = (float*)alloc(2 * 4);
    float* sw     = (float*)alloc((size_t)NSE * 4);
    float* bw     = (float*)alloc((size_t)NBIP * 4);
    float* lbuf   = (float*)alloc((size_t)NBIP * 4);
    float* ssum   = (float*)alloc((size_t)NCLU * 4);
    float* sn_agg = (float*)alloc((size_t)NCLU * LATENT * 4);
    float* sn_enc = (float*)alloc((size_t)NCLU * (LATENT - EMB) * 4);
    float* snA = (float*)alloc((size_t)NCLU * LATENT * 4);
    float* snB = (float*)alloc((size_t)NCLU * LATENT * 4);
    float* seA = (float*)alloc((size_t)NSE * LATENT * 4);
    float* seB = (float*)alloc((size_t)NSE * LATENT * 4);
    float* ndA = (float*)alloc((size_t)NNODE * LATENT * 4);
    float* ndB = (float*)alloc((size_t)NNODE * LATENT * 4);
    float* edA = (float*)alloc((size_t)NEDGE * LATENT * 4);
    float* edB = (float*)alloc((size_t)NEDGE * LATENT * 4);
    float* msg   = (float*)alloc((size_t)NNODE * LATENT * 4);
    float* attn  = (float*)alloc((size_t)NNODE * LATENT * 4);
    float* smsg  = (float*)alloc((size_t)NCLU * LATENT * 4);
    float* sattn = (float*)alloc((size_t)NCLU * LATENT * 4);
    const size_t EPAD = (size_t)cdiv(NEDGE, 64) * 64;
    half_t* XH = (half_t*)alloc(EPAD * (3 * LATENT) * sizeof(half_t));
    half_t* HB = (half_t*)alloc(EPAD * HIDDEN * sizeof(half_t));

    // --- convert all MLP weights to transposed f16 -------------------------
    auto convW = [&](const MlpP& m, const WtH& w, int K, int H, int O) {
        k_convert_wt<<<gb((size_t)K * H), 256, 0, stream>>>(m.W0, w.W0, K, H);
        k_convert_wt<<<gb((size_t)H * O), 256, 0, stream>>>(m.W1, w.W1, H, O);
    };
    convW(snenc, w_snenc, LATENT, HIDDEN, LATENT - EMB);
    convW(seenc, w_seenc, 2 * LATENT, HIDDEN, LATENT);
    convW(outp, w_out, LATENT, HIDDEN, EMB);
    for (int ci = 0; ci < 2; ++ci) {
        convW(c_edge[ci], w_cedge[ci], 3 * LATENT, HIDDEN, LATENT);
        convW(c_node[ci], w_cnode[ci], 3 * LATENT, HIDDEN, LATENT);
        convW(c_sedge[ci], w_csedge[ci], 3 * LATENT, HIDDEN, LATENT);
        convW(c_snode[ci], w_csnode[ci], 3 * LATENT, HIDDEN, LATENT);
    }

    // --- cluster means (scatter_mean + normalize) --------------------------
    k_zero<<<gb((size_t)NCLU * EMB), 256, 0, stream>>>(means, (size_t)NCLU * EMB);
    k_zero<<<gb(NCLU), 256, 0, stream>>>(cnt, NCLU);
    k_scatter_emb<<<gb((size_t)NNODE * EMB), 256, 0, stream>>>(
        embeddings, clusters, means, cnt, (size_t)NNODE * EMB);
    k_finalize_means<<<gb(NCLU), 256, 0, stream>>>(means, cnt, NCLU);

    // --- super-graph edge weights (sigmoid) --------------------------------
    k_edge_dot<<<gb(NSE), 256, 0, stream>>>(means, means, sg0, sg1, lbuf, NSE);
    k_zero<<<1, 32, 0, stream>>>(stats, 2);
    k_stats<<<gb(NSE), 256, 0, stream>>>(lbuf, NSE, stats);
    k_zero<<<gb(NCLU), 256, 0, stream>>>(ssum, NCLU);
    k_edge_w<<<gb(NSE), 256, 0, stream>>>(lbuf, stats, sup_g, sup_b, 0, sw,
                                          ssum, sg1, NSE);
    k_edge_norm<<<gb(NSE), 256, 0, stream>>>(sw, ssum, sg1, NSE);

    // --- bipartite edge weights (exp) --------------------------------------
    k_edge_dot<<<gb(NBIP), 256, 0, stream>>>(embeddings, means, bsrc, bdst,
                                             lbuf, NBIP);
    k_zero<<<1, 32, 0, stream>>>(stats, 2);
    k_stats<<<gb(NBIP), 256, 0, stream>>>(lbuf, NBIP, stats);
    k_zero<<<gb(NCLU), 256, 0, stream>>>(ssum, NCLU);
    k_edge_w<<<gb(NBIP), 256, 0, stream>>>(lbuf, stats, bip_g, bip_b, 1, bw,
                                           ssum, bdst, NBIP);
    k_edge_norm<<<gb(NBIP), 256, 0, stream>>>(bw, ssum, bdst, NBIP);

    // --- supernode seeds: sn_agg -> encoder -> concat with means -----------
    k_zero<<<gb((size_t)NCLU * LATENT), 256, 0, stream>>>(sn_agg,
                                                          (size_t)NCLU * LATENT);
    k_sgw_add<<<gb((size_t)NBIP * LATENT), 256, 0, stream>>>(
        nodes_in, bsrc, bdst, bw, sn_agg, NBIP, LATENT);
    k_f2h<<<gb((size_t)NCLU * LATENT), 256, 0, stream>>>(sn_agg, XH,
                                                         (size_t)NCLU * LATENT);
    launch_layer<LATENT, HIDDEN>(XH, w_snenc.W0, snenc.b0, snenc.g0, snenc.be0,
                                 nullptr, HB, nullptr, NCLU, 1, stream);
    launch_layer<HIDDEN, LATENT - EMB>(HB, w_snenc.W1, snenc.b1, snenc.g1,
                                       snenc.be1, nullptr, nullptr, sn_enc,
                                       NCLU, 1, stream);
    k_concat2f<<<gb((size_t)NCLU * LATENT), 256, 0, stream>>>(means, sn_enc,
                                                              snA, NCLU);

    // --- superedge encoder --------------------------------------------------
    k_gather2cat1h<<<gb((size_t)NSE * 2 * LATENT), 256, 0, stream>>>(
        snA, sg0, sg1, nullptr, XH, NSE, LATENT, 2 * LATENT);
    launch_layer<2 * LATENT, HIDDEN>(XH, w_seenc.W0, seenc.b0, seenc.g0,
                                     seenc.be0, nullptr, HB, nullptr, NSE, 1,
                                     stream);
    launch_layer<HIDDEN, LATENT>(HB, w_seenc.W1, seenc.b1, seenc.g1, seenc.be1,
                                 nullptr, nullptr, seA, NSE, 1, stream);

    // --- message passing cells ---------------------------------------------
    const float* curN = nodes_in;
    const float* curE = edges_in;
    const float* curS = snA;
    const float* curQ = seA;
    float* nxtN = ndA; float* nxtE = edA; float* nxtS = snB; float* nxtQ = seB;
    for (int ci = 0; ci < 2; ++ci) {
        // node_msg = segsum(edges, g1) + segsum(edges, g0)
        k_zero<<<gb((size_t)NNODE * LATENT), 256, 0, stream>>>(
            msg, (size_t)NNODE * LATENT);
        k_sgw_add<<<gb((size_t)NEDGE * LATENT), 256, 0, stream>>>(
            curE, nullptr, graph1, nullptr, msg, NEDGE, LATENT);
        k_sgw_add<<<gb((size_t)NEDGE * LATENT), 256, 0, stream>>>(
            curE, nullptr, graph0, nullptr, msg, NEDGE, LATENT);
        // attn = segsum(supernodes[bdst]*bw, bsrc)
        k_zero<<<gb((size_t)NNODE * LATENT), 256, 0, stream>>>(
            attn, (size_t)NNODE * LATENT);
        k_sgw_add<<<gb((size_t)NBIP * LATENT), 256, 0, stream>>>(
            curS, bdst, bsrc, bw, attn, NBIP, LATENT);
        // nodes = MLP([nodes|msg|attn]) + nodes
        k_concat3h<<<gb((size_t)NNODE * 3 * LATENT), 256, 0, stream>>>(
            curN, msg, attn, XH, NNODE);
        launch_layer<3 * LATENT, HIDDEN>(XH, w_cnode[ci].W0, c_node[ci].b0,
                                         c_node[ci].g0, c_node[ci].be0, nullptr,
                                         HB, nullptr, NNODE, 1, stream);
        launch_layer<HIDDEN, LATENT>(HB, w_cnode[ci].W1, c_node[ci].b1,
                                     c_node[ci].g1, c_node[ci].be1, curN,
                                     nullptr, nxtN, NNODE, 1, stream);
        const float* newN = nxtN;
        // edges = MLP([nodes[g0]|nodes[g1]|edges]) + edges
        k_gather2cat1h<<<gb((size_t)NEDGE * 3 * LATENT), 256, 0, stream>>>(
            newN, graph0, graph1, curE, XH, NEDGE, LATENT, 3 * LATENT);
        launch_layer<3 * LATENT, HIDDEN>(XH, w_cedge[ci].W0, c_edge[ci].b0,
                                         c_edge[ci].g0, c_edge[ci].be0, nullptr,
                                         HB, nullptr, NEDGE, 1, stream);
        launch_layer<HIDDEN, LATENT>(HB, w_cedge[ci].W1, c_edge[ci].b1,
                                     c_edge[ci].g1, c_edge[ci].be1, curE,
                                     nullptr, nxtE, NEDGE, 1, stream);
        const float* newE = nxtE;
        // sn_msg = segsum(superedges*sw, sg1) + segsum(superedges*sw, sg0)
        k_zero<<<gb((size_t)NCLU * LATENT), 256, 0, stream>>>(
            smsg, (size_t)NCLU * LATENT);
        k_sgw_add<<<gb((size_t)NSE * LATENT), 256, 0, stream>>>(
            curQ, nullptr, sg1, sw, smsg, NSE, LATENT);
        k_sgw_add<<<gb((size_t)NSE * LATENT), 256, 0, stream>>>(
            curQ, nullptr, sg0, sw, smsg, NSE, LATENT);
        // sattn = segsum(nodes[bsrc]*bw, bdst)  (uses NEW nodes)
        k_zero<<<gb((size_t)NCLU * LATENT), 256, 0, stream>>>(
            sattn, (size_t)NCLU * LATENT);
        k_sgw_add<<<gb((size_t)NBIP * LATENT), 256, 0, stream>>>(
            newN, bsrc, bdst, bw, sattn, NBIP, LATENT);
        // supernodes = MLP([sn|sn_msg|sattn]) + sn
        k_concat3h<<<gb((size_t)NCLU * 3 * LATENT), 256, 0, stream>>>(
            curS, smsg, sattn, XH, NCLU);
        launch_layer<3 * LATENT, HIDDEN>(XH, w_csnode[ci].W0, c_snode[ci].b0,
                                         c_snode[ci].g0, c_snode[ci].be0,
                                         nullptr, HB, nullptr, NCLU, 1, stream);
        launch_layer<HIDDEN, LATENT>(HB, w_csnode[ci].W1, c_snode[ci].b1,
                                     c_snode[ci].g1, c_snode[ci].be1, curS,
                                     nullptr, nxtS, NCLU, 1, stream);
        const float* newS = nxtS;
        // superedges = MLP([sn[sg0]|sn[sg1]|se]) + se  (uses NEW supernodes)
        k_gather2cat1h<<<gb((size_t)NSE * 3 * LATENT), 256, 0, stream>>>(
            newS, sg0, sg1, curQ, XH, NSE, LATENT, 3 * LATENT);
        launch_layer<3 * LATENT, HIDDEN>(XH, w_csedge[ci].W0, c_sedge[ci].b0,
                                         c_sedge[ci].g0, c_sedge[ci].be0,
                                         nullptr, HB, nullptr, NSE, 1, stream);
        launch_layer<HIDDEN, LATENT>(HB, w_csedge[ci].W1, c_sedge[ci].b1,
                                     c_sedge[ci].g1, c_sedge[ci].be1, curQ,
                                     nullptr, nxtQ, NSE, 1, stream);
        // rotate ping-pong
        curN = newN; curE = newE; curS = newS; curQ = nxtQ;
        nxtN = ndB; nxtE = edB; nxtS = snA; nxtQ = seA;  // second-iter targets
    }

    // --- output layer + L2 normalize, then clusters ------------------------
    k_f2h<<<gb((size_t)NNODE * LATENT), 256, 0, stream>>>(
        curN, XH, (size_t)NNODE * LATENT);
    launch_layer<LATENT, HIDDEN>(XH, w_out.W0, outp.b0, outp.g0, outp.be0,
                                 nullptr, HB, nullptr, NNODE, 1, stream);
    float* outF = (float*)d_out;
    launch_layer<HIDDEN, EMB>(HB, w_out.W1, outp.b1, outp.g1, outp.be1, nullptr,
                              nullptr, outF, NNODE, 0, stream);
    k_l2norm_rows<<<gb(NNODE), 256, 0, stream>>>(outF, NNODE, EMB);
    k_copy_i32<<<gb(NNODE), 256, 0, stream>>>(
        clusters, ((int*)d_out) + (size_t)NNODE * EMB, NNODE);
}
